// MultiHeadAttention_10024453669491
// MI455X (gfx1250) — matrix operand order repaired and checked by reading, not compiled, not run on hardware
//
#include <hip/hip_runtime.h>
#include <hip/hip_bf16.h>

// ---------------------------------------------------------------------------
// MHA forward for MI455X (gfx1250, wave32, WMMA).
//   1) cvt f32->f16 for q,k,v and all weight matrices
//   2) WMMA GEMM: Qp = q@Wq^T+bq (f16), Kp = k@Wk^T+bk (f16)
//   3) WMMA GEMM (batched over B, transposed-C store): Vp_t[b][n][s] (f16)
//   4) WMMA GEMM (batched over B*H): scores = 0.125 * Qp_h @ Kp_h^T (f32, ws, NT)
//   5) softmax rows (NT read) -> fp32 attn to d_out (NT), f16 probs in-place
//   6) WMMA GEMM (batched over B*H): O = attn_f16 @ Vp_t^T -> merged heads f16
//   7) WMMA GEMM: out = O @ Wo^T + bo (f32) -> d_out
// GEMM core: v_wmma_f32_16x16x32_f16, double-buffered LDS tiles filled with
// GLOBAL_LOAD_ASYNC_TO_LDS_B128 (ASYNCcnt) so DMA overlaps WMMA compute.
// Streaming fp32 score/attn traffic (2x536MB >> 192MB L2) uses NT hints so L2
// is kept for the reused f16 operands.
// ---------------------------------------------------------------------------

typedef __attribute__((ext_vector_type(16))) _Float16 v16h;
typedef __attribute__((ext_vector_type(8)))  _Float16 v8h;
typedef __attribute__((ext_vector_type(8)))  float    v8f;
typedef __attribute__((ext_vector_type(4)))  int      v4i;

typedef __attribute__((address_space(1))) v4i gv4i;  // global int4
typedef __attribute__((address_space(3))) v4i lv4i;  // LDS int4

#define BM 128
#define BN 64
#define BK 32
#define LDS_AS 40   // 32 + 8 halves pad: keeps 16B alignment, avoids bank conflicts
#define LDS_BS 40

#if __has_builtin(__builtin_amdgcn_global_load_async_to_lds_b128)
#define USE_ASYNC_LDS 1
#else
#define USE_ASYNC_LDS 0
#endif

__device__ __forceinline__ void async_cp16(const _Float16* g, _Float16* l) {
#if USE_ASYNC_LDS
  __builtin_amdgcn_global_load_async_to_lds_b128(
      (gv4i*)(void*)g, (lv4i*)(void*)l, 0, 0);
#else
  (void)g; (void)l;
#endif
}

__device__ __forceinline__ void wait_async0() {
#if __has_builtin(__builtin_amdgcn_s_wait_asynccnt)
  __builtin_amdgcn_s_wait_asynccnt(0);
#else
  asm volatile("s_wait_asynccnt 0x0" ::: "memory");
#endif
}

// ---------------------------------------------------------------------------
__global__ void cvt_f32_to_f16(const float* __restrict__ src,
                               _Float16* __restrict__ dst, long n) {
  long i = (long)blockIdx.x * blockDim.x + threadIdx.x;
  long stride = (long)gridDim.x * blockDim.x;
  for (; i < n; i += stride) dst[i] = (_Float16)src[i];
}

// ---------------------------------------------------------------------------
// Batched WMMA GEMM:  C[m,n] = alpha * sum_k A[m,k]*B[n,k] (+ bias[n])
//   A: (M x K) f16, row stride lda, batch offset bb*sAb + hh*sAh (z: bb=z/H, hh=z%H)
//   B: (N x K) f16, row stride ldb, batch offset bb*sBb + hh*sBh
//   C: element (m,n) at coff + m*ldcm + n*ldcn; fp32 if F32OUT else f16
//   NTOUT: use non-temporal stores for C (write-once streaming output)
// Requires: M%128==0 (per batch), N%64==0, K%32==0, row strides multiples of 8.
template <bool F32OUT, bool HAS_BIAS, bool NTOUT>
__global__ __launch_bounds__(256) void wmma_gemm_f16(
    const _Float16* __restrict__ A, const _Float16* __restrict__ B,
    void* __restrict__ Cv, const float* __restrict__ bias, float alpha,
    int M, int N, int K, int lda, int ldb, long ldcm, long ldcn,
    int H, long sAb, long sAh, long sBb, long sBh, long sCb, long sCh) {
  __shared__ __align__(16) _Float16 As[2][BM * LDS_AS];
  __shared__ __align__(16) _Float16 Bs[2][BN * LDS_BS];

  const int z  = blockIdx.z;
  const int bb = z / H;
  const int hh = z % H;
  A += (size_t)bb * sAb + (size_t)hh * sAh;
  B += (size_t)bb * sBb + (size_t)hh * sBh;
  const size_t coff = (size_t)bb * sCb + (size_t)hh * sCh;

  const int tid  = threadIdx.x;
  const int lane = tid & 31;
  const int wave = tid >> 5;
  const int wm   = wave & 3;   // 4 waves along M -> 32-row slabs
  const int wn   = wave >> 2;  // 2 waves along N -> 32-col slabs

  const int rowBase = blockIdx.y * BM;
  const int colBase = blockIdx.x * BN;

  v8f acc[2][2] = {};

  // global->LDS tile mapping (256 threads): A 128x32 (2 thr/row, 16h each),
  // B 64x32 (4 thr/row, 8h each)
  const int arow = tid >> 1;
  const int acol = (tid & 1) * 16;
  const int brow = tid >> 2;
  const int bcol = (tid & 3) * 8;

  const _Float16* aG = A + (size_t)(rowBase + arow) * lda + acol;
  const _Float16* bG = B + (size_t)(colBase + brow) * ldb + bcol;
  _Float16* aL[2] = { &As[0][arow * LDS_AS + acol], &As[1][arow * LDS_AS + acol] };
  _Float16* bL[2] = { &Bs[0][brow * LDS_BS + bcol], &Bs[1][brow * LDS_BS + bcol] };

  const int fm    = lane & 15;  // fragment row/col within 16
  const int khalf = lane >> 4;  // half-lane K split

  auto compute = [&](int buf) {
    v16h afr[2], bfr[2];
#pragma unroll
    for (int i = 0; i < 2; ++i) {
      // A 16x32 f16: lanes 0-15 hold K {0..7,16..23}, lanes 16-31 hold {8..15,24..31}
      const _Float16* ap = &As[buf][(wm * 32 + i * 16 + fm) * LDS_AS];
      v8h lo = *(const v8h*)(ap + khalf * 8);
      v8h hi = *(const v8h*)(ap + 16 + khalf * 8);
#pragma unroll
      for (int e = 0; e < 8; ++e) { afr[i][e] = lo[e]; afr[i][8 + e] = hi[e]; }
    }
#pragma unroll
    for (int j = 0; j < 2; ++j) {
      // B 32x16 f16: lanes 0-15 hold K {0..7,16..23}, lanes 16-31 hold {8..15,24..31}
      const _Float16* bp = &Bs[buf][(wn * 32 + j * 16 + fm) * LDS_BS];
      v8h lo = *(const v8h*)(bp + khalf * 8);
      v8h hi = *(const v8h*)(bp + 16 + khalf * 8);
#pragma unroll
      for (int e = 0; e < 8; ++e) { bfr[j][e] = lo[e]; bfr[j][8 + e] = hi[e]; }
    }
#pragma unroll
    for (int i = 0; i < 2; ++i)
#pragma unroll
      for (int j = 0; j < 2; ++j)
        acc[i][j] = __builtin_amdgcn_wmma_f32_16x16x32_f16(
            false, afr[i], false, bfr[j], (short)0, acc[i][j], false, false);
  };

  const int nk = K / BK;
#if USE_ASYNC_LDS
  // ---- double-buffered async pipeline: stage k+1 while computing k ----
  async_cp16(aG, aL[0]);
  async_cp16(aG + 8, aL[0] + 8);
  async_cp16(bG, bL[0]);
  for (int kt = 0; kt < nk; ++kt) {
    const int cur = kt & 1;
    wait_async0();        // this wave's stage kt landed in LDS
    __syncthreads();      // all waves staged kt; all done computing kt-1
    if (kt + 1 < nk) {    // async-fill the other buffer; overlaps compute below
      const _Float16* ag = aG + (size_t)(kt + 1) * BK;
      const _Float16* bg = bG + (size_t)(kt + 1) * BK;
      async_cp16(ag, aL[cur ^ 1]);
      async_cp16(ag + 8, aL[cur ^ 1] + 8);
      async_cp16(bg, bL[cur ^ 1]);
    }
    compute(cur);
  }
#else
  // ---- fallback: synchronous single-buffer staging ----
  for (int kt = 0; kt < nk; ++kt) {
    __syncthreads();
    const _Float16* ag = aG + (size_t)kt * BK;
    *(v8h*)aL[0]       = *(const v8h*)ag;
    *(v8h*)(aL[0] + 8) = *(const v8h*)(ag + 8);
    *(v8h*)bL[0]       = *(const v8h*)(bG + (size_t)kt * BK);
    __syncthreads();
    compute(0);
  }
#endif

  // epilogue: C/D layout = 8 VGPRs, lanes 0-15 rows 0-7, lanes 16-31 rows 8-15
  float*    Cf = (float*)Cv;
  _Float16* Ch = (_Float16*)Cv;
  const int nl = lane & 15;
  const int rh = lane >> 4;
#pragma unroll
  for (int i = 0; i < 2; ++i) {
    const int mt = rowBase + wm * 32 + i * 16 + rh * 8;
#pragma unroll
    for (int j = 0; j < 2; ++j) {
      const int nt = colBase + wn * 32 + j * 16 + nl;
      float bv = 0.0f;
      if constexpr (HAS_BIAS) bv = bias[nt];
      size_t idx = coff + (size_t)mt * ldcm + (size_t)nt * ldcn;
#pragma unroll
      for (int v = 0; v < 8; ++v) {
        const float val = alpha * acc[i][j][v] + bv;
        if constexpr (F32OUT) {
          if constexpr (NTOUT) __builtin_nontemporal_store(val, &Cf[idx]);
          else                 Cf[idx] = val;
        } else {
          if constexpr (NTOUT) __builtin_nontemporal_store((_Float16)val, &Ch[idx]);
          else                 Ch[idx] = (_Float16)val;
        }
        idx += (size_t)ldcm;
      }
    }
  }
}

// ---------------------------------------------------------------------------
// Row softmax over S=2048 columns. NT-loads the streaming fp32 scores
// (last use), NT-stores fp32 probs to attn_out (never re-read), and writes an
// f16 copy *in-place* into the first half of this row's own fp32 storage with
// regular temporal hint (re-read soon by the attn@V GEMM).
// Safe: every thread finishes reading the row before the reduction barriers.
__global__ __launch_bounds__(256) void softmax_rows(
    float* __restrict__ scores, float* __restrict__ attn_out, int S) {
  __shared__ float red[256];
  const size_t row = blockIdx.x;
  float*    r    = scores + row * (size_t)S;
  _Float16* r16  = (_Float16*)scores + row * (size_t)(2 * S);  // aliases own row only
  float*    outp = attn_out + row * (size_t)S;
  const int tid = threadIdx.x;

  float vals[8];
  float mx = -3.402823466e38f;
#pragma unroll
  for (int i = 0; i < 8; ++i) {
    vals[i] = __builtin_nontemporal_load(&r[tid + i * 256]);
    mx = fmaxf(mx, vals[i]);
  }
  red[tid] = mx; __syncthreads();
  for (int s = 128; s > 0; s >>= 1) {
    if (tid < s) red[tid] = fmaxf(red[tid], red[tid + s]);
    __syncthreads();
  }
  mx = red[0]; __syncthreads();

  float sum = 0.f;
#pragma unroll
  for (int i = 0; i < 8; ++i) { vals[i] = __expf(vals[i] - mx); sum += vals[i]; }
  red[tid] = sum; __syncthreads();
  for (int s = 128; s > 0; s >>= 1) {
    if (tid < s) red[tid] += red[tid + s];
    __syncthreads();
  }
  const float inv = 1.0f / red[0];

#pragma unroll
  for (int i = 0; i < 8; ++i) {
    const float p = vals[i] * inv;
    __builtin_nontemporal_store(p, &outp[tid + i * 256]);  // fp32 attn output
    r16[tid + i * 256] = (_Float16)p;  // f16 probs for attn@V (lda = 2*S)
  }
}

// ---------------------------------------------------------------------------
extern "C" void kernel_launch(void* const* d_in, const int* in_sizes, int n_in,
                              void* d_out, int out_size, void* d_ws, size_t ws_size,
                              hipStream_t stream) {
  (void)in_sizes; (void)n_in; (void)out_size; (void)ws_size;
  constexpr int  Bz = 2, S = 2048, D = 1024, H = 16, DK = 64;
  constexpr long MD = (long)Bz * S;  // 4096 rows

  const float* q  = (const float*)d_in[0];
  const float* k  = (const float*)d_in[1];
  const float* v  = (const float*)d_in[2];
  const float* wq = (const float*)d_in[3];
  const float* bq = (const float*)d_in[4];
  const float* wk = (const float*)d_in[5];
  const float* bk = (const float*)d_in[6];
  const float* wv = (const float*)d_in[7];
  const float* bv = (const float*)d_in[8];
  const float* wo = (const float*)d_in[9];
  const float* bo = (const float*)d_in[10];

  float* out_p  = (float*)d_out;                   // [B,S,D]
  float* attn_p = (float*)d_out + (size_t)MD * D;  // [B,H,S,S]

  // workspace layout
  char* ws = (char*)d_ws;
  size_t off = 0;
  auto alloc = [&](size_t bytes) -> void* {
    void* p = ws + off;
    off = (off + bytes + 255) & ~(size_t)255;
    return p;
  };
  _Float16* Qf  = (_Float16*)alloc((size_t)MD * D * 2);
  _Float16* Kf  = (_Float16*)alloc((size_t)MD * D * 2);
  _Float16* Vf  = (_Float16*)alloc((size_t)MD * D * 2);
  _Float16* Wqf = (_Float16*)alloc((size_t)D * D * 2);
  _Float16* Wkf = (_Float16*)alloc((size_t)D * D * 2);
  _Float16* Wvf = (_Float16*)alloc((size_t)D * D * 2);
  _Float16* Wof = (_Float16*)alloc((size_t)D * D * 2);
  _Float16* Qp  = (_Float16*)alloc((size_t)MD * D * 2);
  _Float16* Kp  = (_Float16*)alloc((size_t)MD * D * 2);
  _Float16* Vpt = (_Float16*)alloc((size_t)MD * D * 2);  // [B][D][S] transposed
  _Float16* Om  = (_Float16*)alloc((size_t)MD * D * 2);  // merged heads [B*S, D]
  float*    Sc  = (float*)alloc((size_t)Bz * H * S * S * 4);

  // 1) conversions
  cvt_f32_to_f16<<<2048, 256, 0, stream>>>(q,  Qf,  MD * D);
  cvt_f32_to_f16<<<2048, 256, 0, stream>>>(k,  Kf,  MD * D);
  cvt_f32_to_f16<<<2048, 256, 0, stream>>>(v,  Vf,  MD * D);
  cvt_f32_to_f16<<<1024, 256, 0, stream>>>(wq, Wqf, (long)D * D);
  cvt_f32_to_f16<<<1024, 256, 0, stream>>>(wk, Wkf, (long)D * D);
  cvt_f32_to_f16<<<1024, 256, 0, stream>>>(wv, Wvf, (long)D * D);
  cvt_f32_to_f16<<<1024, 256, 0, stream>>>(wo, Wof, (long)D * D);

  // 2) Q/K projections: [4096,1024] = Xf @ W^T + b, f16 out, row-major
  {
    dim3 grid(D / BN, MD / BM, 1);
    wmma_gemm_f16<false, true, false><<<grid, 256, 0, stream>>>(
        Qf, Wqf, Qp, bq, 1.0f,
        (int)MD, D, D, D, D, (long)D, 1L, 1, 0, 0, 0, 0, 0, 0);
    wmma_gemm_f16<false, true, false><<<grid, 256, 0, stream>>>(
        Kf, Wkf, Kp, bk, 1.0f,
        (int)MD, D, D, D, D, (long)D, 1L, 1, 0, 0, 0, 0, 0, 0);
  }

  // 3) V projection, stored transposed per batch: Vpt[b][n][s] (ldcm=1, ldcn=S)
  {
    dim3 grid(D / BN, S / BM, Bz);
    wmma_gemm_f16<false, true, false><<<grid, 256, 0, stream>>>(
        Vf, Wvf, Vpt, bv, 1.0f,
        S, D, D, D, D, 1L, (long)S,
        1, (long)S * D, 0, 0, 0, (long)D * S, 0);
  }

  // 4) scores[b,h] = 0.125 * Qp_h @ Kp_h^T (f32, batched over B*H, NT stores)
  {
    dim3 grid(S / BN, S / BM, Bz * H);
    wmma_gemm_f16<true, false, true><<<grid, 256, 0, stream>>>(
        Qp, Kp, Sc, nullptr, 0.125f,
        S, S, DK, D, D, (long)S, 1L,
        H, (long)S * D, (long)DK, (long)S * D, (long)DK,
        (long)H * S * S, (long)S * S);
  }

  // 5) softmax rows -> fp32 attn (d_out, NT) + f16 probs aliased in-place (lda = 2S)
  softmax_rows<<<Bz * H * S, 256, 0, stream>>>(Sc, attn_p, S);

  // 6) O[b,h] = attn_f16 @ Vpt_h^T -> merged-head f16 [B*S, D]
  {
    dim3 grid(DK / BN, S / BM, Bz * H);
    wmma_gemm_f16<false, false, false><<<grid, 256, 0, stream>>>(
        (const _Float16*)Sc, Vpt, Om, nullptr, 1.0f,
        S, DK, S, 2 * S, S, (long)D, 1L,
        H, (long)H * S * 2 * S, (long)S * 2 * S,  // attn_f16 strides (elements)
        (long)D * S, (long)DK * S,                // Vpt strides
        (long)S * D, (long)DK);                   // Om offsets: b*S*D + h*64
  }

  // 7) out = Om @ Wo^T + bo -> fp32 d_out
  {
    dim3 grid(D / BN, MD / BM, 1);
    wmma_gemm_f16<true, true, false><<<grid, 256, 0, stream>>>(
        Om, Wof, out_p, bo, 1.0f,
        (int)MD, D, D, D, D, (long)D, 1L, 1, 0, 0, 0, 0, 0, 0);
  }
}
